// GRUD_cell_20529943674895
// MI455X (gfx1250) — compile-verified, matrix operand order checked by reading, not run on hardware
//
#include <hip/hip_runtime.h>
#include <hip/hip_bf16.h>
#include <cstdint>

// Problem dims (match reference)
#define Bm 256
#define Tm 256
#define Fm 128
#define Hm 512
#define H3 (3*Hm)
#define KI 768      // 2F + H  = W_ih K-dim
#define KA 1280     // KI + H  = fused [W_ih | W_hh] K-dim

typedef __attribute__((ext_vector_type(16))) __bf16 v16bf;
typedef __attribute__((ext_vector_type(8)))  __bf16 v8bf;
typedef __attribute__((ext_vector_type(4)))  __bf16 v4bf;
typedef __attribute__((ext_vector_type(8)))  float  v8f;

union V16 { v16bf v; v8bf h[2]; };

__device__ __forceinline__ float fast_sigmoid(float x) {
    return __builtin_amdgcn_rcpf(1.f + __expf(-x));            // v_exp + v_rcp
}
__device__ __forceinline__ float fast_tanh(float x) {
    // tanh(x) = 1 - 2/(exp(2x)+1); saturates correctly at +/-inf
    return 1.f - 2.f * __builtin_amdgcn_rcpf(1.f + __expf(2.f * x));
}

// ---- WMMA fragment loader --------------------------------------------------
// 16-bit A/B 16x32 layout (ISA 7.12.2): lanes 0-15 hold rows 0-15 with
// K = {0..7} in VGPR0-3 and K = {16..23} in VGPR4-7; lanes 16-31 hold the
// K = {8..15} / {24..31} halves.  Each lane reads two 16-byte chunks which are
// *already* in fragment order -> reinterpret directly, no element shuffling.
__device__ __forceinline__ v16bf load_frag_bf16(const __bf16* base,
                                                int row0, int ld, int k, int lane) {
    int r    = row0 + (lane & 15);
    int koff = (lane >> 4) << 3;                 // 0 or 8
    const __bf16* p = base + (long)r * ld + k + koff;
    V16 t;
    t.h[0] = *(const v8bf*)(p);                  // K koff..koff+7
    t.h[1] = *(const v8bf*)(p + 16);             // K koff+16..koff+23
    return t.v;
}

#define WMMA_BF16(a, b, c) \
    __builtin_amdgcn_wmma_f32_16x16x32_bf16(false, (a), false, (b), (short)0, (c), false, false)

// ---- One-time prep: fp32 -> bf16 weights (fused W_ih|W_hh) and deltas ------
__global__ void grud_prep(const float* __restrict__ Wh,
                          const float* __restrict__ Wih,
                          const float* __restrict__ Whh,
                          const float* __restrict__ deltas,
                          __bf16* __restrict__ Whb,
                          __bf16* __restrict__ Wcat,
                          __bf16* __restrict__ Dbuf) {
    long stride = (long)gridDim.x * blockDim.x;
    long tid = (long)blockIdx.x * blockDim.x + threadIdx.x;

    for (long i = tid; i < Hm * Fm; i += stride)
        Whb[i] = (__bf16)Wh[i];

    long n2 = (long)H3 * KA;
    for (long i = tid; i < n2; i += stride) {
        int j = (int)(i / KA), k = (int)(i % KA);
        float v = (k < KI) ? Wih[(long)j * KI + k] : Whh[(long)j * Hm + (k - KI)];
        Wcat[i] = (__bf16)v;
    }

    long n3 = (long)Bm * Tm * Fm;                // divisible by 4
    for (long i = tid * 4; i < n3; i += stride * 4) {
        float4 d = *(const float4*)(deltas + i);
        v4bf o;
        o[0] = (__bf16)d.x; o[1] = (__bf16)d.y;
        o[2] = (__bf16)d.z; o[3] = (__bf16)d.w;
        *(v4bf*)(Dbuf + i) = o;
    }
}

__global__ void grud_init_h(float* __restrict__ hbuf) {
    int i = blockIdx.x * blockDim.x + threadIdx.x;
    if (i < Bm * Hm) hbuf[i] = 0.f;
}

// ---- Per-step elementwise: gamma_x (diag decay), x_rep, m -> Abuf ----------
__global__ void grud_phase1_elem(const float* __restrict__ values,
                                 const float* __restrict__ masks,
                                 const float* __restrict__ deltas,
                                 const float* __restrict__ emp,
                                 const float* __restrict__ xlocf,
                                 const float* __restrict__ wx,
                                 const float* __restrict__ bx,
                                 __bf16* __restrict__ Abuf, int t) {
    int i = blockIdx.x * blockDim.x + threadIdx.x;   // over B*F
    if (i >= Bm * Fm) return;
    int b = i / Fm, f = i % Fm;
    long idx = ((long)b * Tm + t) * Fm + f;
    float d  = deltas[idx];
    float gx = __expf(-fmaxf(d * wx[f] + bx[f], 0.f));
    float xh = gx * xlocf[idx] + (1.f - gx) * emp[f];
    float m  = masks[idx];
    float xr = m * values[idx] + (1.f - m) * xh;
    Abuf[(long)b * KA + f]           = (__bf16)xr;   // x_rep slice [0,128)
    Abuf[(long)b * KA + Fm + Hm + f] = (__bf16)m;    // m slice [640,768)
}

// ---- Per-step decay GEMM: gamma_h = exp(-relu(d@Wh^T+bh)); h *= gamma_h ----
__global__ void grud_phase1_gemm(const __bf16* __restrict__ Dbuf,
                                 const __bf16* __restrict__ Whb,
                                 const float* __restrict__ bh,
                                 float* __restrict__ hbuf,
                                 __bf16* __restrict__ Abuf, int t) {
    int lane = threadIdx.x & 31;
    int wave = threadIdx.x >> 5;
    int tile = blockIdx.x * 4 + wave;     // 16 row-tiles x 32 col-tiles = 512
    int bt = tile >> 5, jt = tile & 31;
    int b0 = bt * 16, j0 = jt * 16;

    const __bf16* Drow = Dbuf + (long)t * Fm;   // row b lives at b*(Tm*Fm)+t*Fm

    v8f acc = {};
#pragma unroll
    for (int k = 0; k < Fm; k += 32) {
        v16bf a  = load_frag_bf16(Drow, b0, Tm * Fm, k, lane);
        v16bf bw = load_frag_bf16(Whb, j0, Fm, k, lane);
        acc = WMMA_BF16(a, bw, acc);
    }
    // C layout: lane = col (lane&15), VGPR r -> row r (+8 for upper lane half)
    int j    = j0 + (lane & 15);
    int half = lane >> 4;
    float bj = bh[j];
#pragma unroll
    for (int r = 0; r < 8; ++r) {
        int b    = b0 + r + 8 * half;
        float g  = __expf(-fmaxf(acc[r] + bj, 0.f));
        float hd = hbuf[(long)b * Hm + j] * g;
        hbuf[(long)b * Hm + j] = hd;                  // fp32 recurrent state
        __bf16 hb = (__bf16)hd;
        Abuf[(long)b * KA + Fm + j] = hb;             // h slice for W_ih
        Abuf[(long)b * KA + KI + j] = hb;             // h slice for W_hh
    }
}

// ---- Per-step gate GEMM + fused GRU epilogue -------------------------------
// Each wave: 32 batch rows x 16 hidden cols.  Two A-fragments share three
// B-fragments (r-rows, z-rows, n-rows of Wcat) -> 6 WMMAs per 10 b128 loads.
// Four logical accumulators per row-half: r, z over full K; i_n over W_ih K,
// h_n over W_hh K, so tanh(i_n + r*h_n) fuses in-wave in the epilogue.
__global__ void grud_phase2(const __bf16* __restrict__ Abuf,
                            const __bf16* __restrict__ Wcat,
                            const float* __restrict__ b_ih,
                            const float* __restrict__ b_hh,
                            float* __restrict__ hbuf,
                            float* __restrict__ out, int t, int last) {
    int lane = threadIdx.x & 31;
    int wave = threadIdx.x >> 5;
    int tile = blockIdx.x * 4 + wave;     // 8 row-pairs x 32 col-tiles = 256
    int bt = tile >> 5, jt = tile & 31;
    int b0 = bt * 32, j0 = jt * 16;

    v8f ar[2]  = { {}, {} };
    v8f az[2]  = { {}, {} };
    v8f an1[2] = { {}, {} };
    v8f an2[2] = { {}, {} };

    for (int k = 0; k < KI; k += 32) {                // W_ih region of Wcat
        v16bf a0 = load_frag_bf16(Abuf, b0,      KA, k, lane);
        v16bf a1 = load_frag_bf16(Abuf, b0 + 16, KA, k, lane);
        v16bf br = load_frag_bf16(Wcat, j0,          KA, k, lane);
        v16bf bz = load_frag_bf16(Wcat, Hm + j0,     KA, k, lane);
        v16bf bn = load_frag_bf16(Wcat, 2 * Hm + j0, KA, k, lane);
        ar[0]  = WMMA_BF16(a0, br, ar[0]);   ar[1]  = WMMA_BF16(a1, br, ar[1]);
        az[0]  = WMMA_BF16(a0, bz, az[0]);   az[1]  = WMMA_BF16(a1, bz, az[1]);
        an1[0] = WMMA_BF16(a0, bn, an1[0]);  an1[1] = WMMA_BF16(a1, bn, an1[1]);
    }
    for (int k = KI; k < KA; k += 32) {               // W_hh region of Wcat
        v16bf a0 = load_frag_bf16(Abuf, b0,      KA, k, lane);
        v16bf a1 = load_frag_bf16(Abuf, b0 + 16, KA, k, lane);
        v16bf br = load_frag_bf16(Wcat, j0,          KA, k, lane);
        v16bf bz = load_frag_bf16(Wcat, Hm + j0,     KA, k, lane);
        v16bf bn = load_frag_bf16(Wcat, 2 * Hm + j0, KA, k, lane);
        ar[0]  = WMMA_BF16(a0, br, ar[0]);   ar[1]  = WMMA_BF16(a1, br, ar[1]);
        az[0]  = WMMA_BF16(a0, bz, az[0]);   az[1]  = WMMA_BF16(a1, bz, az[1]);
        an2[0] = WMMA_BF16(a0, bn, an2[0]);  an2[1] = WMMA_BF16(a1, bn, an2[1]);
    }

    int j    = j0 + (lane & 15);
    int half = lane >> 4;
    float bir = b_ih[j]          + b_hh[j];
    float biz = b_ih[Hm + j]     + b_hh[Hm + j];
    float bin = b_ih[2 * Hm + j];
    float bhn = b_hh[2 * Hm + j];
#pragma unroll
    for (int ht = 0; ht < 2; ++ht) {
        int bb0 = b0 + 16 * ht;
#pragma unroll
        for (int r = 0; r < 8; ++r) {
            int b = bb0 + r + 8 * half;
            float rr = fast_sigmoid(ar[ht][r] + bir);
            float zz = fast_sigmoid(az[ht][r] + biz);
            float nn = fast_tanh(an1[ht][r] + bin + rr * (an2[ht][r] + bhn));
            float hp = hbuf[(long)b * Hm + j];        // decayed h (fp32)
            float hv = (1.f - zz) * nn + zz * hp;
            hbuf[(long)b * Hm + j] = hv;
            out[((long)b * Tm + t) * Hm + j] = hv;    // hs[b, t, :]
            if (last) out[(long)Bm * Tm * Hm + (long)b * Hm + j] = hv;  // h_last
        }
    }
}

// ---- Host driver -----------------------------------------------------------
extern "C" void kernel_launch(void* const* d_in, const int* in_sizes, int n_in,
                              void* d_out, int out_size, void* d_ws, size_t ws_size,
                              hipStream_t stream) {
    const float* values = (const float*)d_in[0];
    const float* masks  = (const float*)d_in[1];
    const float* deltas = (const float*)d_in[2];
    const float* emp    = (const float*)d_in[3];
    const float* xlocf  = (const float*)d_in[4];
    const float* wx     = (const float*)d_in[5];
    const float* bx     = (const float*)d_in[6];
    const float* Wh     = (const float*)d_in[7];
    const float* bh     = (const float*)d_in[8];
    const float* Wih    = (const float*)d_in[9];
    const float* Whh    = (const float*)d_in[10];
    const float* bih    = (const float*)d_in[11];
    const float* bhh    = (const float*)d_in[12];
    float* out = (float*)d_out;

    char* ws = (char*)d_ws;
    // ws layout (bytes): Whb 128K | Wcat 3.75M | Abuf 640K | hbuf 512K | Dbuf 16M
    __bf16* Whb  = (__bf16*)(ws);
    __bf16* Wcat = (__bf16*)(ws + 131072);
    __bf16* Abuf = (__bf16*)(ws + 131072 + 3932160);
    float*  hbuf = (float*)(ws + 131072 + 3932160 + 655360);
    __bf16* Dbuf = (__bf16*)(ws + 131072 + 3932160 + 655360 + 524288);

    grud_prep<<<1024, 256, 0, stream>>>(Wh, Wih, Whh, deltas, Whb, Wcat, Dbuf);
    grud_init_h<<<(Bm * Hm) / 256, 256, 0, stream>>>(hbuf);

    for (int t = 0; t < Tm; ++t) {
        grud_phase1_elem<<<(Bm * Fm) / 256, 256, 0, stream>>>(
            values, masks, deltas, emp, xlocf, wx, bx, Abuf, t);
        grud_phase1_gemm<<<128, 128, 0, stream>>>(Dbuf, Whb, bh, hbuf, Abuf, t);
        grud_phase2<<<64, 128, 0, stream>>>(Abuf, Wcat, bih, bhh, hbuf, out, t,
                                            (t == Tm - 1) ? 1 : 0);
    }
}
